// GFocalHead_7241314861769
// MI455X (gfx1250) — compile-verified, matrix operand order checked by reading, not run on hardware
//
#include <hip/hip_runtime.h>

typedef __attribute__((ext_vector_type(16))) __bf16 v16bf;
typedef __attribute__((ext_vector_type(8)))  __bf16 v8bf;
typedef __attribute__((ext_vector_type(8)))  float  v8f;

#define TROWS 4
#define TCOLS 32
#define TPOS  128      // positions per block tile
#define KPAD  104      // padded bf16 row length (208 B, 16B-aligned rows)

static __device__ __forceinline__ __bf16 f2bf(float f) {
    unsigned u = __float_as_uint(f);
    u += 0x7FFFu + ((u >> 16) & 1u);               // round-to-nearest-even
    unsigned short h = (unsigned short)(u >> 16);
    __bf16 r;
    __builtin_memcpy(&r, &h, 2);
    return r;
}

// CDNA5 async global->LDS DMA of one dword per lane (tracked with ASYNCcnt).
static __device__ __forceinline__ void async_ld_b32(const float* g, float* lds_generic) {
    unsigned lds = (unsigned)(uintptr_t)lds_generic;           // low 32 bits = LDS offset
    unsigned long long ga = (unsigned long long)(uintptr_t)g;
    asm volatile("global_load_async_to_lds_b32 %0, %1, off"
                 :: "v"(lds), "v"(ga) : "memory");
}
static __device__ __forceinline__ void async_wait0() {
    asm volatile("s_wait_asynccnt 0" ::: "memory");
}

// Fused depthwise 3x3 (+bias, ReLU) -> pointwise 96x96 (+bias, ReLU) via WMMA bf16.
// x: [B,96,H,W] f32, y: [B,96,H,W] f32
__global__ __launch_bounds__(256) void dwpw_kernel(
    const float* __restrict__ x,  const float* __restrict__ dww,
    const float* __restrict__ dwb, const float* __restrict__ pww,
    const float* __restrict__ pwb, float* __restrict__ y,
    int H, int W, int tilesX, int tilesY)
{
    __shared__ __align__(16) __bf16 sW[96 * KPAD];     // pointwise weights [m][k]
    __shared__ __align__(16) __bf16 sAct[TPOS * KPAD]; // activations [pos][k]
    __shared__ __align__(16) float sDwW[96 * 9];
    __shared__ __align__(16) float sDwB[96];
    __shared__ __align__(16) float sPwB[96];

    const int tid = threadIdx.x;

    // Small fp32 constants: async DMA straight into LDS (overlaps with bf16 convert below)
    for (int i = tid; i < 96 * 9; i += 256) async_ld_b32(dww + i, &sDwW[i]);
    if (tid < 96) {
        async_ld_b32(dwb + tid, &sDwB[tid]);
        async_ld_b32(pwb + tid, &sPwB[tid]);
    }

    // Pointwise weights: load + convert fp32 -> bf16 on VALU while DMA is in flight
    for (int i = tid; i < 96 * 96; i += 256) {
        int m = i / 96, kk = i - m * 96;
        sW[m * KPAD + kk] = f2bf(pww[i]);
    }

    int blk = blockIdx.x;
    const int tX = blk % tilesX; blk /= tilesX;
    const int tY = blk % tilesY; blk /= tilesY;
    const int b  = blk;
    const int h0 = tY * TROWS, w0 = tX * TCOLS;

    async_wait0();
    __syncthreads();

    // ---- depthwise 3x3 + ReLU (VALU) -> LDS bf16 ----
    {
        const int pos = tid & (TPOS - 1);
        const int pr = pos >> 5, pc = pos & 31;
        const int hh = h0 + pr, ww = w0 + pc;
        const bool inb = (hh < H) && (ww < W);
        const float* xb = x + (size_t)b * 96 * H * W;
        for (int c = (tid >> 7); c < 96; c += 2) {
            float acc = 0.f;
            if (inb) {
                const float* xc = xb + (size_t)c * H * W;
                const float* wk = &sDwW[c * 9];
                acc = sDwB[c];
                __builtin_prefetch(xc + 2 * (size_t)H * W + (size_t)hh * W + ww, 0, 1);
                #pragma unroll
                for (int dy = -1; dy <= 1; ++dy) {
                    const int hy = hh + dy;
                    if ((unsigned)hy < (unsigned)H) {
                        #pragma unroll
                        for (int dx = -1; dx <= 1; ++dx) {
                            const int wx = ww + dx;
                            if ((unsigned)wx < (unsigned)W)
                                acc += wk[(dy + 1) * 3 + (dx + 1)] * xc[(size_t)hy * W + wx];
                        }
                    }
                }
                acc = acc > 0.f ? acc : 0.f;
            }
            sAct[pos * KPAD + c] = f2bf(acc);
        }
    }
    __syncthreads();

    // ---- pointwise 96x96 GEMM via v_wmma_f32_16x16x32_bf16 ----
    {
        const int lane = tid & 31;
        const int ln = lane & 15;
        const int halfid = lane >> 4;
        const int nt = tid >> 5;            // wave id = N-tile (8 x 16 positions)
        const int pos = nt * 16 + ln;
        const __bf16* actRow = &sAct[pos * KPAD];
        const int hh = h0 + (pos >> 5), ww = w0 + (pos & 31);
        const bool inb = (hh < H) && (ww < W);
        #pragma unroll
        for (int mt = 0; mt < 6; ++mt) {
            v8f acc = {0.f, 0.f, 0.f, 0.f, 0.f, 0.f, 0.f, 0.f};
            const __bf16* wRow = &sW[(mt * 16 + ln) * KPAD];
            #pragma unroll
            for (int kb = 0; kb < 3; ++kb) {
                const int klo = kb * 32 + halfid * 8;
                v8bf a0 = *(const v8bf*)(wRow + klo);
                v8bf a1 = *(const v8bf*)(wRow + klo + 16);
                v8bf b0 = *(const v8bf*)(actRow + klo);
                v8bf b1 = *(const v8bf*)(actRow + klo + 16);
                v16bf A, Bm;
                #pragma unroll
                for (int i = 0; i < 8; ++i) { A[i] = a0[i]; A[i + 8] = a1[i]; Bm[i] = b0[i]; Bm[i + 8] = b1[i]; }
                acc = __builtin_amdgcn_wmma_f32_16x16x32_bf16(false, A, false, Bm, (short)0, acc, false, false);
            }
            if (inb) {
                #pragma unroll
                for (int v = 0; v < 8; ++v) {
                    const int co = mt * 16 + halfid * 8 + v;
                    float val = acc[v] + sPwB[co];
                    val = val > 0.f ? val : 0.f;
                    y[(((size_t)b * 96 + co) * H + hh) * W + ww] = val;
                }
            }
        }
    }
}

// Head 1x1 conv: 112x96 GEMM via WMMA, +bias, writes position-major [B*H*W][112].
__global__ __launch_bounds__(256) void head_kernel(
    const float* __restrict__ x, const float* __restrict__ hw,
    const float* __restrict__ hb, float* __restrict__ out,
    int H, int W, int tilesX, int tilesY)
{
    __shared__ __align__(16) __bf16 sW[112 * KPAD];
    __shared__ __align__(16) __bf16 sAct[TPOS * KPAD];
    __shared__ __align__(16) float sB[112];

    const int tid = threadIdx.x;

    if (tid < 112) async_ld_b32(hb + tid, &sB[tid]);   // ASYNCcnt DMA for bias

    for (int i = tid; i < 112 * 96; i += 256) {
        int m = i / 96, kk = i - m * 96;
        sW[m * KPAD + kk] = f2bf(hw[i]);
    }

    int blk = blockIdx.x;
    const int tX = blk % tilesX; blk /= tilesX;
    const int tY = blk % tilesY; blk /= tilesY;
    const int b  = blk;
    const int h0 = tY * TROWS, w0 = tX * TCOLS;

    {
        const int pos = tid & (TPOS - 1);
        const int pr = pos >> 5, pc = pos & 31;
        const int hh = h0 + pr, ww = w0 + pc;
        const bool inb = (hh < H) && (ww < W);
        const float* xb = x + (size_t)b * 96 * H * W;
        for (int c = (tid >> 7); c < 96; c += 2) {
            float v = inb ? xb[((size_t)c * H + hh) * W + ww] : 0.f;
            sAct[pos * KPAD + c] = f2bf(v);
        }
    }
    async_wait0();
    __syncthreads();

    const int lane = tid & 31;
    const int ln = lane & 15;
    const int halfid = lane >> 4;
    const int nt = tid >> 5;
    const int pos = nt * 16 + ln;
    const __bf16* actRow = &sAct[pos * KPAD];
    const int hh = h0 + (pos >> 5), ww = w0 + (pos & 31);
    const bool inb = (hh < H) && (ww < W);
    const int HWl = H * W;
    #pragma unroll
    for (int mt = 0; mt < 7; ++mt) {
        v8f acc = {0.f, 0.f, 0.f, 0.f, 0.f, 0.f, 0.f, 0.f};
        const __bf16* wRow = &sW[(mt * 16 + ln) * KPAD];
        #pragma unroll
        for (int kb = 0; kb < 3; ++kb) {
            const int klo = kb * 32 + halfid * 8;
            v8bf a0 = *(const v8bf*)(wRow + klo);
            v8bf a1 = *(const v8bf*)(wRow + klo + 16);
            v8bf b0 = *(const v8bf*)(actRow + klo);
            v8bf b1 = *(const v8bf*)(actRow + klo + 16);
            v16bf A, Bm;
            #pragma unroll
            for (int i = 0; i < 8; ++i) { A[i] = a0[i]; A[i + 8] = a1[i]; Bm[i] = b0[i]; Bm[i + 8] = b1[i]; }
            acc = __builtin_amdgcn_wmma_f32_16x16x32_bf16(false, A, false, Bm, (short)0, acc, false, false);
        }
        if (inb) {
            const int m0 = mt * 16 + halfid * 8;
            float* orow = out + ((size_t)b * HWl + (size_t)hh * W + ww) * 112 + m0;
            float4 lo  = make_float4(acc[0] + sB[m0 + 0], acc[1] + sB[m0 + 1],
                                     acc[2] + sB[m0 + 2], acc[3] + sB[m0 + 3]);
            float4 hi4 = make_float4(acc[4] + sB[m0 + 4], acc[5] + sB[m0 + 5],
                                     acc[6] + sB[m0 + 6], acc[7] + sB[m0 + 7]);
            *(float4*)(orow)     = lo;
            *(float4*)(orow + 4) = hi4;
        }
    }
}

// DFL decode + sigmoid; head: [B*HW][112] -> out rows [B, Ntot, 84]
__global__ __launch_bounds__(256) void post_kernel(
    const float* __restrict__ head, float* __restrict__ out,
    int HW, int W, float stride, int nOff, int Ntot, int total)
{
    int idx = blockIdx.x * 256 + threadIdx.x;
    if (idx >= total) return;
    const int b = idx / HW;
    const int p = idx - b * HW;
    const int py = p / W, px = p - py * W;
    const float* hrow = head + (size_t)idx * 112;
    float* orow = out + ((size_t)b * Ntot + nOff + p) * 84;
    const float cx = (px + 0.5f) * stride;
    const float cy = (py + 0.5f) * stride;
    float d[4];
    #pragma unroll
    for (int j = 0; j < 4; ++j) {
        const float* r = hrow + 80 + j * 8;
        float m = r[0];
        #pragma unroll
        for (int i = 1; i < 8; ++i) m = fmaxf(m, r[i]);
        float s = 0.f, num = 0.f;
        #pragma unroll
        for (int i = 0; i < 8; ++i) {
            float e = __expf(r[i] - m);
            s += e; num += e * (float)i;
        }
        d[j] = (num / s) * stride;
    }
    orow[0] = cx - d[0];
    orow[1] = cy - d[1];
    orow[2] = cx + d[2];
    orow[3] = cy + d[3];
    for (int c = 0; c < 80; ++c)
        orow[4 + c] = 1.f / (1.f + __expf(-hrow[c]));
}

extern "C" void kernel_launch(void* const* d_in, const int* in_sizes, int n_in,
                              void* d_out, int out_size, void* d_ws, size_t ws_size,
                              hipStream_t stream) {
    (void)in_sizes; (void)n_in; (void)out_size; (void)ws_size;
    const int B = 32;
    const int Hs[3] = {80, 40, 20};
    const float Sd[3] = {8.f, 16.f, 32.f};
    const int Noff[3] = {0, 6400, 8000};
    const int Ntot = 8400;

    // ws ping-pong: both regions sized for worst case B*112*6400 floats (~92 MB each)
    float* bufA = (float*)d_ws;
    float* bufB = bufA + (size_t)B * 112 * 6400;
    float* outp = (float*)d_out;

    for (int k = 0; k < 3; ++k) {
        const int H = Hs[k], W = Hs[k], HW = H * W;
        const int tX = (W + TCOLS - 1) / TCOLS, tY = (H + TROWS - 1) / TROWS;
        dim3 grid(B * tX * tY), blkd(256);
        const int p = 3 + k * 10; // params flattened after x0,x1,x2 in dict order
        const float* xin = (const float*)d_in[k];

        // stack 0: x -> bufA (NCHW)
        dwpw_kernel<<<grid, blkd, 0, stream>>>(
            xin, (const float*)d_in[p + 0], (const float*)d_in[p + 1],
            (const float*)d_in[p + 2], (const float*)d_in[p + 3],
            bufA, H, W, tX, tY);
        // stack 1: bufA -> bufB (NCHW)
        dwpw_kernel<<<grid, blkd, 0, stream>>>(
            bufA, (const float*)d_in[p + 4], (const float*)d_in[p + 5],
            (const float*)d_in[p + 6], (const float*)d_in[p + 7],
            bufB, H, W, tX, tY);
        // head: bufB -> bufA (position-major [B*HW][112])
        head_kernel<<<grid, blkd, 0, stream>>>(
            bufB, (const float*)d_in[p + 8], (const float*)d_in[p + 9],
            bufA, H, W, tX, tY);
        // postprocess -> d_out
        const int tot = B * HW;
        post_kernel<<<(tot + 255) / 256, 256, 0, stream>>>(
            bufA, outp, HW, W, Sd[k], Noff[k], Ntot, tot);
    }
}
